// Attn_54812372631855
// MI455X (gfx1250) — matrix operand order repaired and checked by reading, not compiled, not run on hardware
//
#include <hip/hip_runtime.h>
#include <hip/hip_fp16.h>

#define DM    1024
#define NSEQ  2048
#define BATCH 4
#define NH    16
#define DH    64

typedef __attribute__((ext_vector_type(16))) _Float16 v16h;
typedef __attribute__((ext_vector_type(8)))  _Float16 v8h;
typedef __attribute__((ext_vector_type(4)))  _Float16 v4h;
typedef __attribute__((ext_vector_type(8)))  float    v8f;
typedef __attribute__((ext_vector_type(4)))  int      v4i;

static __device__ __forceinline__ v8h  ld8h (const _Float16* p){ return *(const v8h*)p;  }
static __device__ __forceinline__ v16h ld16h(const _Float16* p){ return *(const v16h*)p; }
static __device__ __forceinline__ v16h cat8(v8h lo, v8h hi){
  return __builtin_shufflevector(lo, hi, 0,1,2,3,4,5,6,7,8,9,10,11,12,13,14,15);
}
static __device__ __forceinline__ v8f wmma16(v16h a, v16h b, v8f c){
  // D = A(16x32 f16) * B(32x16 f16) + C(16x16 f32)
  return __builtin_amdgcn_wmma_f32_16x16x32_f16(false, a, false, b, (short)0, c, false, false);
}

// -------- CDNA5 async global->LDS path (ASYNCcnt), with sync fallback --------
#if defined(__HIP_DEVICE_COMPILE__) && __has_builtin(__builtin_amdgcn_global_load_async_to_lds_b128)
#define ATTN_ASYNC 1
#else
#define ATTN_ASYNC 0
#endif

#if ATTN_ASYNC
typedef __attribute__((address_space(1))) v4i GV4;   // global int4*
typedef __attribute__((address_space(3))) v4i LV4;   // LDS int4*
static __device__ __forceinline__ void async_cp16(const _Float16* g, _Float16* l){
  v4i* gg = (v4i*)g;          // generic, typed
  v4i* ll = (v4i*)l;
  __builtin_amdgcn_global_load_async_to_lds_b128((GV4*)gg, (LV4*)ll, 0, 0);
}
#endif

static __device__ __forceinline__ void wait_async(){
#if ATTN_ASYNC
#if __has_builtin(__builtin_amdgcn_s_wait_asynccnt)
  __builtin_amdgcn_s_wait_asynccnt(0);
#else
  asm volatile("s_wait_asynccnt 0x0" ::: "memory");
#endif
#endif
}

// ---------------------------------------------------------------- convert f32 -> f16
__global__ void cvt_f32_f16(const float* __restrict__ src, _Float16* __restrict__ dst, int n){
  for (int i = (blockIdx.x*blockDim.x + threadIdx.x)*4; i < n; i += gridDim.x*blockDim.x*4){
    float4 v = *(const float4*)(src + i);
    v4h h; h[0]=(_Float16)v.x; h[1]=(_Float16)v.y; h[2]=(_Float16)v.z; h[3]=(_Float16)v.w;
    *(v4h*)(dst + i) = h;
  }
}

// ---------------------------------------------------------------- fused QKV projection
// grid: (M/128, DM/64, 3), block 128. Each wave: 32x64 tile (2 A-frags x 4 B-frags).
// z=0 -> Q (scaled 1/8) [b,h,n,64] ; z=1 -> K [b,h,n,64] ; z=2 -> V transposed [b,h,64,n]
__global__ __launch_bounds__(128) void qkv_kernel(
    const _Float16* __restrict__ xh,
    const _Float16* __restrict__ Wqh, const _Float16* __restrict__ Wkh, const _Float16* __restrict__ Wvh,
    const float* __restrict__ bq, const float* __restrict__ bk, const float* __restrict__ bv,
    _Float16* __restrict__ Qh, _Float16* __restrict__ Kh, _Float16* __restrict__ Vt)
{
  const int lane = threadIdx.x & 31, wave = threadIdx.x >> 5;
  const int ln = lane & 15, lh = lane >> 4;
  const int m0 = blockIdx.x*128 + wave*32;
  const int h  = blockIdx.y;
  const int z  = blockIdx.z;
  const _Float16* W    = (z==0) ? Wqh : (z==1) ? Wkh : Wvh;
  const float*    bias = (z==0) ? bq  : (z==1) ? bk  : bv;

  const _Float16* aR0 = xh + (size_t)(m0 + ln)*DM;
  const _Float16* aR1 = xh + (size_t)(m0 + 16 + ln)*DM;
  const _Float16* bR0 = W + (size_t)(h*64      + ln)*DM;
  const _Float16* bR1 = W + (size_t)(h*64 + 16 + ln)*DM;
  const _Float16* bR2 = W + (size_t)(h*64 + 32 + ln)*DM;
  const _Float16* bR3 = W + (size_t)(h*64 + 48 + ln)*DM;

  v8f c[2][4] = {};
  #pragma unroll 2
  for (int k0 = 0; k0 < DM; k0 += 32){
    v16h a0 = ld16h(aR0 + k0 + lh*16);
    v16h a1 = ld16h(aR1 + k0 + lh*16);
    v16h b0 = ld16h(bR0 + k0 + lh*16);
    v16h b1 = ld16h(bR1 + k0 + lh*16);
    v16h b2 = ld16h(bR2 + k0 + lh*16);
    v16h b3 = ld16h(bR3 + k0 + lh*16);
    c[0][0] = wmma16(a0, b0, c[0][0]);  c[1][0] = wmma16(a1, b0, c[1][0]);
    c[0][1] = wmma16(a0, b1, c[0][1]);  c[1][1] = wmma16(a1, b1, c[1][1]);
    c[0][2] = wmma16(a0, b2, c[0][2]);  c[1][2] = wmma16(a1, b2, c[1][2]);
    c[0][3] = wmma16(a0, b3, c[0][3]);  c[1][3] = wmma16(a1, b3, c[1][3]);
  }

  const int b  = m0 / NSEQ;
  const int bh = b*NH + h;
  #pragma unroll
  for (int j = 0; j < 4; ++j){
    const int d = j*16 + ln;
    const float bb = bias[h*64 + d];
    #pragma unroll
    for (int f = 0; f < 2; ++f){
      const int nbase = (m0 % NSEQ) + f*16 + lh*8;    // 8 consecutive seq rows / lane
      if (z == 2){                                    // V transposed: packed b128 store
        v8h pv;
        #pragma unroll
        for (int i = 0; i < 8; ++i) pv[i] = (_Float16)(c[f][j][i] + bb);
        *(v8h*)(Vt + ((size_t)bh*DH + d)*NSEQ + nbase) = pv;
      } else {
        _Float16* dst = (z==0) ? Qh : Kh;
        const float scale = (z==0) ? 0.125f : 1.0f;   // 1/sqrt(64) folded into Q
        #pragma unroll
        for (int i = 0; i < 8; ++i)
          dst[((size_t)bh*NSEQ + nbase + i)*DH + d] = (_Float16)((c[f][j][i] + bb)*scale);
      }
    }
  }
}

// ---------------------------------------------------------------- flash attention (causal)
// K tile 32x64 halves (128B/row), V tile (transposed) 64x32 halves (64B/row); 32B per thread each.
static __device__ __forceinline__ void stage_kv(
    const _Float16* __restrict__ Kbase, const _Float16* __restrict__ Vbase,
    int k0, _Float16* sK, _Float16* sV, int t)
{
  const int r = t >> 2, p = t & 3;
  const _Float16* ks = Kbase + (size_t)(k0 + r)*DH + p*16;
  _Float16* kd = sK + r*64 + p*16;
  const int dd = t >> 1, ph = t & 1;
  const _Float16* vs = Vbase + (size_t)dd*NSEQ + k0 + ph*16;
  _Float16* vd = sV + dd*32 + ph*16;
#if ATTN_ASYNC
  async_cp16(ks, kd);      async_cp16(ks + 8, kd + 8);
  async_cp16(vs, vd);      async_cp16(vs + 8, vd + 8);
#else
  *(v8h*)kd       = ld8h(ks);     *(v8h*)(kd + 8) = ld8h(ks + 8);
  *(v8h*)vd       = ld8h(vs);     *(v8h*)(vd + 8) = ld8h(vs + 8);
#endif
}

// grid: (NSEQ/64, BATCH*NH) block 128. Wave w owns q rows [qblk*64+w*16, +16).
__global__ __launch_bounds__(128) void attn_kernel(
    const _Float16* __restrict__ Qh, const _Float16* __restrict__ Kh,
    const _Float16* __restrict__ Vt, _Float16* __restrict__ Ah)
{
  __shared__ __align__(32) _Float16 sK[2][32*64];   // [buf][key][d]
  __shared__ __align__(32) _Float16 sV[2][64*32];   // [buf][d][key]

  const int lane = threadIdx.x & 31, wave = threadIdx.x >> 5;
  const int ln = lane & 15, lh = lane >> 4;
  const int qblk = blockIdx.x;
  const int bh = blockIdx.y;
  const int b = bh >> 4, h = bh & 15;
  const int q0 = qblk*64 + wave*16;
  const int qrow = q0 + ln;

  const _Float16* Qbase = Qh + (size_t)bh*NSEQ*DH;
  const _Float16* Kbase = Kh + (size_t)bh*NSEQ*DH;
  const _Float16* Vbase = Vt + (size_t)bh*DH*NSEQ;

  // Q^T B-fragments (col = q, K-dim = d), resident across the key loop
  const v16h qb0 = cat8(ld8h(Qbase + (size_t)qrow*DH + lh*8), ld8h(Qbase + (size_t)qrow*DH + 16 + lh*8));        // d 0..31
  const v16h qb1 = cat8(ld8h(Qbase + (size_t)qrow*DH + 32 + lh*8), ld8h(Qbase + (size_t)qrow*DH + 48 + lh*8));   // d 32..63

  v8f o0 = {}, o1 = {}, o2 = {}, o3 = {};
  float m_run = -1e30f, l_run = 0.0f;

  const int nkb = 2*(qblk + 1);                  // 32-key blocks up to causal limit
  stage_kv(Kbase, Vbase, 0, sK[0], sV[0], threadIdx.x);   // prime the pipeline

  for (int kb = 0; kb < nkb; ++kb){
    const int k0 = kb*32;
    wait_async();            // my async tile transfers have landed in LDS
    __syncthreads();         // everyone's have; prev buffer fully consumed
    const int cur = kb & 1;
    if (kb + 1 < nkb)
      stage_kv(Kbase, Vbase, k0 + 32, sK[cur^1], sV[cur^1], threadIdx.x);

    // S^T = K_tile * Q^T : C-layout (lane=q col, VGPRs=keys) feeds P A-frag directly
    v8f st0 = {}, st1 = {};
    {
      const _Float16* kr0 = sK[cur] + ln*64;          // keys 0..15 of block
      const _Float16* kr1 = sK[cur] + (16 + ln)*64;   // keys 16..31
      v16h a;
      a = cat8(ld8h(kr0 + lh*8),      ld8h(kr0 + 16 + lh*8)); st0 = wmma16(a, qb0, st0);
      a = cat8(ld8h(kr0 + 32 + lh*8), ld8h(kr0 + 48 + lh*8)); st0 = wmma16(a, qb1, st0);
      a = cat8(ld8h(kr1 + lh*8),      ld8h(kr1 + 16 + lh*8)); st1 = wmma16(a, qb0, st1);
      a = cat8(ld8h(kr1 + 32 + lh*8), ld8h(kr1 + 48 + lh*8)); st1 = wmma16(a, qb1, st1);
    }

    // causal mask (key > q)
    #pragma unroll
    for (int i = 0; i < 8; ++i){
      if (k0 + i + 8*lh > qrow)      st0[i] = -1e30f;
      if (k0 + 16 + i + 8*lh > qrow) st1[i] = -1e30f;
    }

    // online softmax: per-q stats in lane = q (dup across halves); one xor-16 reduce
    float mb = -1e30f;
    #pragma unroll
    for (int i = 0; i < 8; ++i){ mb = fmaxf(mb, st0[i]); mb = fmaxf(mb, st1[i]); }
    mb = fmaxf(mb, __shfl_xor(mb, 16, 32));
    const float m_new = fmaxf(m_run, mb);
    const float alpha = __expf(m_run - m_new);

    v16h pa;                                     // P A-fragment, built fully in-lane
    float s = 0.0f;
    #pragma unroll
    for (int i = 0; i < 8; ++i){
      const float e0 = __expf(st0[i] - m_new);
      const float e1 = __expf(st1[i] - m_new);
      s += e0 + e1;
      pa[i]     = (_Float16)e0;
      pa[8 + i] = (_Float16)e1;
    }
    s += __shfl_xor(s, 16, 32);
    l_run = l_run*alpha + s;
    m_run = m_new;

    // rescale O by per-row alpha (broadcast from lane = row)
    #pragma unroll
    for (int i = 0; i < 8; ++i){
      const float af = __shfl(alpha, lh*8 + i, 32);
      o0[i] *= af; o1[i] *= af; o2[i] *= af; o3[i] *= af;
    }
    // O += P * V  (B-frag: col = v-dim, keys contiguous from transposed V tile)
    const _Float16* vb = sV[cur];
    o0 = wmma16(pa, cat8(ld8h(vb + ln*32      + lh*8), ld8h(vb + ln*32      + 16 + lh*8)), o0);
    o1 = wmma16(pa, cat8(ld8h(vb + (16+ln)*32 + lh*8), ld8h(vb + (16+ln)*32 + 16 + lh*8)), o1);
    o2 = wmma16(pa, cat8(ld8h(vb + (32+ln)*32 + lh*8), ld8h(vb + (32+ln)*32 + 16 + lh*8)), o2);
    o3 = wmma16(pa, cat8(ld8h(vb + (48+ln)*32 + lh*8), ld8h(vb + (48+ln)*32 + 16 + lh*8)), o3);
  }

  const float linv = 1.0f / l_run;
  _Float16* dst = Ah + (size_t)b*NSEQ*DM + (size_t)h*DH;
  #pragma unroll
  for (int i = 0; i < 8; ++i){
    const float li = __shfl(linv, lh*8 + i, 32);
    const int q = q0 + i + 8*lh;
    _Float16* row = dst + (size_t)q*DM;
    row[ln]      = (_Float16)(o0[i]*li);
    row[16 + ln] = (_Float16)(o1[i]*li);
    row[32 + ln] = (_Float16)(o2[i]*li);
    row[48 + ln] = (_Float16)(o3[i]*li);
  }
}

// ---------------------------------------------------------------- output projection (f32 out)
// grid: (M/128, DM/64), block 128. Each wave: 32x64 tile.
__global__ __launch_bounds__(128) void proj_kernel(
    const _Float16* __restrict__ Ah, const _Float16* __restrict__ Woh, float* __restrict__ out)
{
  const int lane = threadIdx.x & 31, wave = threadIdx.x >> 5;
  const int ln = lane & 15, lh = lane >> 4;
  const int m0 = blockIdx.x*128 + wave*32;
  const int oB = blockIdx.y*64;
  const _Float16* aR0 = Ah  + (size_t)(m0 + ln)*DM;
  const _Float16* aR1 = Ah  + (size_t)(m0 + 16 + ln)*DM;
  const _Float16* bR0 = Woh + (size_t)(oB      + ln)*DM;
  const _Float16* bR1 = Woh + (size_t)(oB + 16 + ln)*DM;
  const _Float16* bR2 = Woh + (size_t)(oB + 32 + ln)*DM;
  const _Float16* bR3 = Woh + (size_t)(oB + 48 + ln)*DM;

  v8f c[2][4] = {};
  #pragma unroll 2
  for (int k0 = 0; k0 < DM; k0 += 32){
    v16h a0 = ld16h(aR0 + k0 + lh*16);
    v16h a1 = ld16h(aR1 + k0 + lh*16);
    v16h b0 = ld16h(bR0 + k0 + lh*16);
    v16h b1 = ld16h(bR1 + k0 + lh*16);
    v16h b2 = ld16h(bR2 + k0 + lh*16);
    v16h b3 = ld16h(bR3 + k0 + lh*16);
    c[0][0] = wmma16(a0, b0, c[0][0]);  c[1][0] = wmma16(a1, b0, c[1][0]);
    c[0][1] = wmma16(a0, b1, c[0][1]);  c[1][1] = wmma16(a1, b1, c[1][1]);
    c[0][2] = wmma16(a0, b2, c[0][2]);  c[1][2] = wmma16(a1, b2, c[1][2]);
    c[0][3] = wmma16(a0, b3, c[0][3]);  c[1][3] = wmma16(a1, b3, c[1][3]);
  }
  #pragma unroll
  for (int j = 0; j < 4; ++j)
    #pragma unroll
    for (int f = 0; f < 2; ++f)
      #pragma unroll
      for (int i = 0; i < 8; ++i)
        out[(size_t)(m0 + f*16 + i + 8*lh)*DM + oB + j*16 + ln] = c[f][j][i];
}

// ---------------------------------------------------------------- host side
extern "C" void kernel_launch(void* const* d_in, const int* in_sizes, int n_in,
                              void* d_out, int out_size, void* d_ws, size_t ws_size,
                              hipStream_t stream) {
  const float* x  = (const float*)d_in[0];
  const float* Wq = (const float*)d_in[1];
  const float* bq = (const float*)d_in[2];
  const float* Wk = (const float*)d_in[3];
  const float* bk = (const float*)d_in[4];
  const float* Wv = (const float*)d_in[5];
  const float* bv = (const float*)d_in[6];
  const float* Wo = (const float*)d_in[7];
  float* out = (float*)d_out;

  const size_t NX = (size_t)BATCH*NSEQ*DM;   // 8,388,608
  const size_t NW = (size_t)DM*DM;           // 1,048,576
  _Float16* xh  = (_Float16*)d_ws;           // total f16 workspace ~92 MB
  _Float16* Wqh = xh  + NX;
  _Float16* Wkh = Wqh + NW;
  _Float16* Wvh = Wkh + NW;
  _Float16* Woh = Wvh + NW;
  _Float16* Qh  = Woh + NW;                  // [b,h,n,64]  (pre-scaled 1/8)
  _Float16* Kh  = Qh  + NX;                  // [b,h,n,64]
  _Float16* Vt  = Kh  + NX;                  // [b,h,64,n]
  _Float16* Ah  = Vt  + NX;                  // [b,n,h*64]

  cvt_f32_f16<<<4096, 256, 0, stream>>>(x,  xh,  (int)NX);
  cvt_f32_f16<<<1024, 256, 0, stream>>>(Wq, Wqh, (int)NW);
  cvt_f32_f16<<<1024, 256, 0, stream>>>(Wk, Wkh, (int)NW);
  cvt_f32_f16<<<1024, 256, 0, stream>>>(Wv, Wvh, (int)NW);
  cvt_f32_f16<<<1024, 256, 0, stream>>>(Wo, Woh, (int)NW);

  qkv_kernel<<<dim3(BATCH*NSEQ/128, DM/64, 3), 128, 0, stream>>>(
      xh, Wqh, Wkh, Wvh, bq, bk, bv, Qh, Kh, Vt);

  attn_kernel<<<dim3(NSEQ/64, BATCH*NH), 128, 0, stream>>>(Qh, Kh, Vt, Ah);

  proj_kernel<<<dim3(BATCH*NSEQ/128, DM/64), 128, 0, stream>>>(Ah, Woh, out);
}
